// CrossModalAttention_69166153335001
// MI455X (gfx1250) — matrix operand order repaired and checked by reading, not compiled, not run on hardware
//
#include <hip/hip_runtime.h>
#include <hip/hip_bf16.h>

// ---------------------------------------------------------------------------
// CDNA5 / gfx1250 cross-modal attention.
// WMMA bf16 GEMMs + Tensor Data Mover (TDM) staging of A-operand strips
// into LDS (320 KB/WGP), flash-style softmax (raw matrix never materialized).
// ---------------------------------------------------------------------------

typedef __bf16 bf16_t;
typedef __attribute__((ext_vector_type(16))) __bf16 v16bf;
typedef __attribute__((ext_vector_type(8)))  __bf16 v8bf;
typedef __attribute__((ext_vector_type(8)))  float  v8f;

typedef unsigned int u32x4 __attribute__((ext_vector_type(4)));
typedef int          i32x4 __attribute__((ext_vector_type(4)));
typedef int          i32x8 __attribute__((ext_vector_type(8)));

__device__ __forceinline__ bf16_t f2bf(float f) {
    union { float f; unsigned u; } c; c.f = f;
    unsigned r = (c.u + 0x7FFFu + ((c.u >> 16) & 1u)) >> 16;   // RNE
    union { unsigned short s; bf16_t b; } o; o.s = (unsigned short)r;
    return o.b;
}

// A-operand fragment (16x32 bf16): lane l<16 holds row M=l, K=0..15;
// lanes 16..31 hold row M=l-16, K=16..31. base already includes
// row*stride + k0 + (lane>>4)*16.
__device__ __forceinline__ v16bf load_frag_rowK(const bf16_t* base) {
    v8bf lo = *reinterpret_cast<const v8bf*>(base);
    v8bf hi = *reinterpret_cast<const v8bf*>(base + 8);
    return __builtin_shufflevector(lo, hi, 0,1,2,3,4,5,6,7,8,9,10,11,12,13,14,15);
}

// B-operand fragment (32x16 bf16): lane l<16 holds N=l, K=0..15 contiguous;
// lanes 16..31 hold N=l-16, K=16..31 contiguous. p = row-major source row.
__device__ __forceinline__ v16bf load_frag_contig(const bf16_t* p) {
    return *reinterpret_cast<const v16bf*>(p);
}

__device__ __forceinline__ v8f wmma_bf16(v16bf a, v16bf b, v8f c) {
    return __builtin_amdgcn_wmma_f32_16x16x32_bf16(false, a, false, b,
                                                   (short)0, c, false, false);
}

// ---------------------------------------------------------------------------
// Tensor Data Mover: async 2-D tile load (bf16 elements) global -> LDS.
// Builds the D# descriptor per CDNA5 ISA ch.8 (group0 128b + group1 256b,
// groups 2/3 zero for a 2-D tensor). All operands must be wave-uniform
// (TDM reads the D# from SGPRs). Tracked by TENSORcnt.
// ---------------------------------------------------------------------------
__device__ __forceinline__ void
tdm_load_2d_bf16(const void* gptr, unsigned ldsByteOff,
                 unsigned tileRows, unsigned tileCols,
                 unsigned tensorRows, unsigned tensorCols,
                 unsigned rowStrideElems) {
    unsigned long long ga = (unsigned long long)gptr;
    u32x4 g0;
    g0[0] = 1u;                                    // count=1, user descriptor
    g0[1] = ldsByteOff;                            // lds_addr (bytes)
    g0[2] = (unsigned)(ga & 0xFFFFFFFFu);          // global_addr[31:0]
    g0[3] = (unsigned)((ga >> 32) & 0x1FFFFFFu)    // global_addr[56:32]
          | (2u << 30);                            // type = 2 ("image")
    i32x8 g1;
    g1[0] = (int)(1u << 16);                       // data_size=1 (2B), mask=0
    g1[1] = (int)((tensorCols & 0xFFFFu) << 16);   // tensor_dim0 lo16
    g1[2] = (int)(((tensorCols >> 16) & 0xFFFFu)
          | ((tensorRows & 0xFFFFu) << 16));       // dim0 hi | tensor_dim1 lo
    g1[3] = (int)(((tensorRows >> 16) & 0xFFFFu)
          | ((tileCols & 0xFFFFu) << 16));         // dim1 hi | tile_dim0
    g1[4] = (int)(tileRows & 0xFFFFu);             // tile_dim1 (tile_dim2=0)
    g1[5] = (int)rowStrideElems;                   // tensor_dim0_stride lo32
    g1[6] = 0;
    g1[7] = 0;
    i32x4 z4 = {0, 0, 0, 0};
#if defined(__clang_major__) && (__clang_major__ >= 23)
    i32x8 z8 = {0, 0, 0, 0, 0, 0, 0, 0};
    __builtin_amdgcn_tensor_load_to_lds(g0, g1, z4, z4, z8, 0);
#else
    __builtin_amdgcn_tensor_load_to_lds(g0, g1, z4, z4, 0);
#endif
}

// ---------------------------------------------------------------------------
// fp32 -> bf16 conversion (grid-stride)
// ---------------------------------------------------------------------------
__global__ void k_f2b(const float* __restrict__ src, bf16_t* __restrict__ dst,
                      long n) {
    long i = (long)blockIdx.x * blockDim.x + threadIdx.x;
    long stride = (long)gridDim.x * blockDim.x;
    for (; i < n; i += stride) dst[i] = f2bf(src[i]);
}

// ---------------------------------------------------------------------------
// bf16 tiled transpose: in [R x C] -> out [C x R]. R,C multiples of 32.
// ---------------------------------------------------------------------------
__global__ void k_transpose_bf16(const bf16_t* __restrict__ in,
                                 bf16_t* __restrict__ out, int R, int C) {
    __shared__ bf16_t t[32][33];
    int bx = blockIdx.x * 32, by = blockIdx.y * 32;
    int x = bx + threadIdx.x;
    #pragma unroll
    for (int i = 0; i < 32; i += 8)
        t[threadIdx.y + i][threadIdx.x] = in[(size_t)(by + threadIdx.y + i) * C + x];
    __syncthreads();
    x = by + threadIdx.x;
    #pragma unroll
    for (int i = 0; i < 32; i += 8)
        out[(size_t)(bx + threadIdx.y + i) * R + x] = t[threadIdx.x][threadIdx.y + i];
}

// ---------------------------------------------------------------------------
// Pass 1: per-row online softmax stats of raw = Xa · Xbᵀ (Xa[Na x D], Xb[Nb x D])
// stats[2r]=rowmax, stats[2r+1]=sumexp. Block = 8 waves x 16 rows = 128 rows.
// Each wave's 16x1024 A strip is TDM-staged into LDS once, reused for all Nb.
// Dynamic LDS: 8 * 32 KB.
// ---------------------------------------------------------------------------
__global__ __launch_bounds__(256) void
k_rowstats(const bf16_t* __restrict__ Xa, const bf16_t* __restrict__ Xb,
           float* __restrict__ stats, int Na, int Nb, int D) {
    extern __shared__ char smem[];
    const int tid  = threadIdx.x;
    const int lane = tid & 31;
    const int l15  = lane & 15;
    const int hi   = lane >> 4;
    const int waveU = __builtin_amdgcn_readfirstlane(tid >> 5);
    const int rowBase = (blockIdx.x * 8 + waveU) * 16;

    // ---- TDM: stage my 16 x D strip of Xa into LDS ----
    const unsigned stripOff = __builtin_amdgcn_groupstaticsize() + waveU * 32768u;
    tdm_load_2d_bf16(Xa + (size_t)rowBase * D, stripOff,
                     16u, (unsigned)D, (unsigned)Na, (unsigned)D, (unsigned)D);
    __builtin_amdgcn_s_wait_tensorcnt(0);
    __syncthreads();
    const bf16_t* myA = (const bf16_t*)(smem + waveU * 32768u);

    float m[8], s[8];
    #pragma unroll
    for (int v = 0; v < 8; ++v) { m[v] = -3.0e38f; s[v] = 0.0f; }

    const bf16_t* aRow = myA + (size_t)l15 * D + hi * 16;

    for (int j0 = 0; j0 < Nb; j0 += 16) {
        v8f acc = {};
        const bf16_t* bRow = Xb + (size_t)(j0 + l15) * D + hi * 16;
        __builtin_prefetch(bRow, 0, 1);
        for (int k0 = 0; k0 < D; k0 += 32) {
            v16bf a = load_frag_rowK(aRow + k0);          // ds_load from LDS
            v16bf b = load_frag_contig(bRow + k0);        // streamed from global
            acc = wmma_bf16(a, b, acc);
        }
        // C layout: lane column = j0+l15, VGPR v -> row rowBase + hi*8 + v
        #pragma unroll
        for (int v = 0; v < 8; ++v) {
            float c  = acc[v];
            float nm = fmaxf(m[v], c);
            s[v] = s[v] * __expf(m[v] - nm) + __expf(c - nm);
            m[v] = nm;
        }
    }
    // combine across the 16 column-lanes of each half-wave
    #pragma unroll
    for (int off = 1; off <= 8; off <<= 1) {
        #pragma unroll
        for (int v = 0; v < 8; ++v) {
            float mo = __shfl_xor(m[v], off, 32);
            float so = __shfl_xor(s[v], off, 32);
            float nm = fmaxf(m[v], mo);
            s[v] = s[v] * __expf(m[v] - nm) + so * __expf(mo - nm);
            m[v] = nm;
        }
    }
    if (l15 == 0) {
        #pragma unroll
        for (int v = 0; v < 8; ++v) {
            int r = rowBase + hi * 8 + v;
            stats[2 * r]     = m[v];
            stats[2 * r + 1] = s[v];
        }
    }
}

// ---------------------------------------------------------------------------
// Pass 2: Out = softmax_row(Xa·Xbᵀ) · Xb, flash style.
// Grid: x = D/128 slices, y = Na/128. Per wave: 16 rows x 128 out columns.
// A strip TDM-staged in LDS; score tiles re-laid C->A via per-wave LDS bounce.
// Dynamic LDS: 8 * 32 KB (+ 8 KB static P scratch).
// ---------------------------------------------------------------------------
__global__ __launch_bounds__(256) void
k_attend(const bf16_t* __restrict__ Xa, const bf16_t* __restrict__ Xb,
         const bf16_t* __restrict__ XbT, const float* __restrict__ stats,
         float* __restrict__ Out, int Na, int Nb, int D) {
    __shared__ bf16_t pS[8][16 * 32];       // per-wave P tile scratch (static)
    extern __shared__ char smem[];          // A strips (dynamic)
    const int tid  = threadIdx.x;
    const int lane = tid & 31;
    const int l15  = lane & 15;
    const int hi   = lane >> 4;
    const int waveU = __builtin_amdgcn_readfirstlane(tid >> 5);
    const int rowBase = (blockIdx.y * 8 + waveU) * 16;
    const int dBase   = blockIdx.x * 128;

    const unsigned stripOff = __builtin_amdgcn_groupstaticsize() + waveU * 32768u;
    tdm_load_2d_bf16(Xa + (size_t)rowBase * D, stripOff,
                     16u, (unsigned)D, (unsigned)Na, (unsigned)D, (unsigned)D);
    __builtin_amdgcn_s_wait_tensorcnt(0);
    __syncthreads();
    const bf16_t* myA = (const bf16_t*)(smem + waveU * 32768u);

    float mrow[8], invs[8];
    #pragma unroll
    for (int v = 0; v < 8; ++v) {
        int r = rowBase + hi * 8 + v;
        mrow[v] = stats[2 * r];
        invs[v] = 1.0f / stats[2 * r + 1];
    }

    v8f acc[8];
    #pragma unroll
    for (int f = 0; f < 8; ++f) acc[f] = (v8f){};

    const bf16_t* aRow = myA + (size_t)l15 * D + hi * 16;
    bf16_t* myP = pS[waveU];

    for (int j0 = 0; j0 < Nb; j0 += 32) {
        // -- two 16x16 score tiles, exp'd into LDS in A-fragment source order
        #pragma unroll
        for (int jt = 0; jt < 2; ++jt) {
            v8f sc = {};
            const bf16_t* bRow = Xb + (size_t)(j0 + jt * 16 + l15) * D + hi * 16;
            __builtin_prefetch(bRow, 0, 1);
            for (int k0 = 0; k0 < D; k0 += 32) {
                v16bf a = load_frag_rowK(aRow + k0);
                v16bf b = load_frag_contig(bRow + k0);
                sc = wmma_bf16(a, b, sc);
            }
            #pragma unroll
            for (int v = 0; v < 8; ++v) {
                float e = __expf(sc[v] - mrow[v]);
                myP[(hi * 8 + v) * 32 + jt * 16 + l15] = f2bf(e);
            }
        }
        // -- read P back as A-operand (16 rows x K=32 probabilities)
        v16bf pa = load_frag_rowK(myP + l15 * 32 + hi * 16);
        #pragma unroll
        for (int f = 0; f < 8; ++f) {
            const bf16_t* btRow =
                XbT + (size_t)(dBase + f * 16 + l15) * Nb + j0 + hi * 16;
            v16bf bb = load_frag_contig(btRow);
            acc[f] = wmma_bf16(pa, bb, acc[f]);
        }
    }
    // epilogue: scale by 1/sumexp, scatter per C layout
    #pragma unroll
    for (int f = 0; f < 8; ++f) {
        #pragma unroll
        for (int v = 0; v < 8; ++v) {
            int r = rowBase + hi * 8 + v;
            Out[(size_t)r * D + dBase + f * 16 + l15] = acc[f][v] * invs[v];
        }
    }
}

// ---------------------------------------------------------------------------
// colsum[f] += sum_rows tanh( Att·Wᵀ + b )  (WMMA GEMM + tanh + column reduce)
// Att [N x D] bf16 (TDM-staged strip), W [F x D] bf16 (row-major = B-operand).
// Dynamic LDS: 8 * 32 KB.
// ---------------------------------------------------------------------------
__global__ __launch_bounds__(256) void
k_proj_tanh(const bf16_t* __restrict__ Att, const bf16_t* __restrict__ W,
            const float* __restrict__ bias, float* __restrict__ colsum,
            int N, int F, int D) {
    extern __shared__ char smem[];
    const int tid  = threadIdx.x;
    const int lane = tid & 31;
    const int l15  = lane & 15;
    const int hi   = lane >> 4;
    const int waveU = __builtin_amdgcn_readfirstlane(tid >> 5);
    const int rowBase = (blockIdx.x * 8 + waveU) * 16;

    const unsigned stripOff = __builtin_amdgcn_groupstaticsize() + waveU * 32768u;
    tdm_load_2d_bf16(Att + (size_t)rowBase * D, stripOff,
                     16u, (unsigned)D, (unsigned)N, (unsigned)D, (unsigned)D);
    __builtin_amdgcn_s_wait_tensorcnt(0);
    __syncthreads();
    const bf16_t* myA = (const bf16_t*)(smem + waveU * 32768u);

    const bf16_t* aRow = myA + (size_t)l15 * D + hi * 16;

    for (int f0 = 0; f0 < F; f0 += 16) {
        v8f acc = {};
        const bf16_t* wRow = W + (size_t)(f0 + l15) * D + hi * 16;
        __builtin_prefetch(wRow, 0, 1);
        for (int k0 = 0; k0 < D; k0 += 32) {
            v16bf a = load_frag_rowK(aRow + k0);
            v16bf b = load_frag_contig(wRow + k0);
            acc = wmma_bf16(a, b, acc);
        }
        float b = bias[f0 + l15];
        float partial = 0.0f;
        #pragma unroll
        for (int v = 0; v < 8; ++v) partial += tanhf(acc[v] + b);
        partial += __shfl_xor(partial, 16, 32);   // rows 0-7 + rows 8-15
        if (hi == 0) atomicAdd(&colsum[f0 + l15], partial);
    }
}

// ---------------------------------------------------------------------------
// column sums of an fp32 [N x D] matrix (coalesced)
// ---------------------------------------------------------------------------
__global__ void k_colsum(const float* __restrict__ X, float* __restrict__ colsum,
                         int D, int rowsPerBlock) {
    int col = blockIdx.x * blockDim.x + threadIdx.x;
    int r0  = blockIdx.y * rowsPerBlock;
    float s = 0.0f;
    for (int r = 0; r < rowsPerBlock; ++r) s += X[(size_t)(r0 + r) * D + col];
    atomicAdd(&colsum[col], s);
}

// ---------------------------------------------------------------------------
// final combine
// ---------------------------------------------------------------------------
__global__ void k_final(const float* __restrict__ ts1, const float* __restrict__ ts2,
                        const float* __restrict__ cs1, const float* __restrict__ cs2,
                        const float* __restrict__ Wf1, const float* __restrict__ bf1,
                        const float* __restrict__ Wf2, const float* __restrict__ bf2,
                        float* __restrict__ out, int F, int D, float invN) {
    int f = blockIdx.x * blockDim.x + threadIdx.x;
    if (f >= F) return;
    float d1 = 0.0f, d2 = 0.0f;
    for (int d = 0; d < D; ++d) {
        d1 += cs2[d] * invN * Wf1[(size_t)f * D + d];
        d2 += cs1[d] * invN * Wf2[(size_t)f * D + d];
    }
    float fil1 = 1.0f / (1.0f + __expf(-(d1 + bf1[f])));
    float fil2 = 1.0f / (1.0f + __expf(-(d2 + bf2[f])));
    out[f]     = (ts1[f] * invN) * fil1;
    out[F + f] = (ts2[f] * invN) * fil2;
}

// ---------------------------------------------------------------------------
extern "C" void kernel_launch(void* const* d_in, const int* in_sizes, int n_in,
                              void* d_out, int out_size, void* d_ws, size_t ws_size,
                              hipStream_t stream) {
    constexpr int N1 = 8192, N2 = 8192, D = 1024, F = 800;
    constexpr size_t STRIP_LDS = 8u * 32768u;   // 8 waves x 32 KB A strips

    const float* m1  = (const float*)d_in[0];
    const float* m2  = (const float*)d_in[1];
    const float* Wf1 = (const float*)d_in[2];
    const float* bf1 = (const float*)d_in[3];
    const float* Wf2 = (const float*)d_in[4];
    const float* bf2 = (const float*)d_in[5];
    const float* W1  = (const float*)d_in[6];
    const float* b1  = (const float*)d_in[7];
    const float* W2  = (const float*)d_in[8];
    const float* b2  = (const float*)d_in[9];
    float* out = (float*)d_out;

    // ---- workspace carve (bump allocator, 256B aligned) ----
    char* w = (char*)d_ws;
    auto carve = [&](size_t bytes) -> void* {
        void* p = (void*)w;
        w += (bytes + 255) & ~(size_t)255;
        return p;
    };
    bf16_t* m1b   = (bf16_t*)carve((size_t)N1 * D * 2);
    bf16_t* m2b   = (bf16_t*)carve((size_t)N2 * D * 2);
    bf16_t* m1bT  = (bf16_t*)carve((size_t)N1 * D * 2);   // [D x N1]
    bf16_t* m2bT  = (bf16_t*)carve((size_t)N2 * D * 2);   // [D x N2]
    bf16_t* W1b   = (bf16_t*)carve((size_t)F * D * 2);
    bf16_t* W2b   = (bf16_t*)carve((size_t)F * D * 2);
    float*  stats1 = (float*)carve((size_t)N2 * 2 * 4);   // rows of m2·m1ᵀ
    float*  stats2 = (float*)carve((size_t)N1 * 2 * 4);   // rows of m1·m2ᵀ
    float*  att1   = (float*)carve((size_t)N2 * D * 4);   // attended_m1
    float*  att2   = (float*)carve((size_t)N1 * D * 4);   // attended_m2
    bf16_t* att1b  = (bf16_t*)carve((size_t)N2 * D * 2);
    bf16_t* att2b  = (bf16_t*)carve((size_t)N1 * D * 2);
    float*  cs1    = (float*)carve((size_t)D * 4);
    float*  cs2    = (float*)carve((size_t)D * 4);
    float*  ts1    = (float*)carve((size_t)F * 4);
    float*  ts2    = (float*)carve((size_t)F * 4);

    // ---- stage 0: bf16 conversions + transposes ----
    k_f2b<<<2048, 256, 0, stream>>>(m1, m1b, (long)N1 * D);
    k_f2b<<<2048, 256, 0, stream>>>(m2, m2b, (long)N2 * D);
    k_f2b<<<1024, 256, 0, stream>>>(W1, W1b, (long)F * D);
    k_f2b<<<1024, 256, 0, stream>>>(W2, W2b, (long)F * D);
    k_transpose_bf16<<<dim3(D / 32, N1 / 32), dim3(32, 8), 0, stream>>>(m1b, m1bT, N1, D);
    k_transpose_bf16<<<dim3(D / 32, N2 / 32), dim3(32, 8), 0, stream>>>(m2b, m2bT, N2, D);

    // ---- stage 1: softmax stats (both orientations of raw) ----
    k_rowstats<<<N1 / 128, 256, STRIP_LDS, stream>>>(m1b, m2b, stats2, N1, N2, D);
    k_rowstats<<<N2 / 128, 256, STRIP_LDS, stream>>>(m2b, m1b, stats1, N2, N1, D);

    // ---- stage 2: attended outputs, flash style ----
    k_attend<<<dim3(D / 128, N1 / 128), 256, STRIP_LDS, stream>>>(
        m1b, m2b, m2bT, stats2, att2, N1, N2, D);
    k_attend<<<dim3(D / 128, N2 / 128), 256, STRIP_LDS, stream>>>(
        m2b, m1b, m1bT, stats1, att1, N2, N1, D);

    // ---- stage 3: reductions + projections ----
    hipMemsetAsync(cs1, 0, (size_t)D * 4, stream);
    hipMemsetAsync(cs2, 0, (size_t)D * 4, stream);
    hipMemsetAsync(ts1, 0, (size_t)F * 4, stream);
    hipMemsetAsync(ts2, 0, (size_t)F * 4, stream);

    k_colsum<<<dim3(D / 256, N2 / 256), 256, 0, stream>>>(att1, cs1, D, 256);
    k_colsum<<<dim3(D / 256, N1 / 256), 256, 0, stream>>>(att2, cs2, D, 256);

    k_f2b<<<2048, 256, 0, stream>>>(att1, att1b, (long)N2 * D);
    k_f2b<<<2048, 256, 0, stream>>>(att2, att2b, (long)N1 * D);

    k_proj_tanh<<<N2 / 128, 256, STRIP_LDS, stream>>>(att1b, W1b, b1, ts1, N2, F, D);
    k_proj_tanh<<<N1 / 128, 256, STRIP_LDS, stream>>>(att2b, W2b, b2, ts2, N1, F, D);

    // ---- stage 4: combine ----
    k_final<<<(F + 255) / 256, 256, 0, stream>>>(ts1, ts2, cs1, cs2,
                                                 Wf1, bf1, Wf2, bf2,
                                                 out, F, D, 1.0f / 8192.0f);
}
